// MultiHeadLatentAttention_18193481465896
// MI455X (gfx1250) — matrix operand order repaired and checked by reading, not compiled, not run on hardware
//
#include <hip/hip_runtime.h>
#include <hip/hip_bf16.h>

// ---------------------------------------------------------------------------
// MLA (multi-head latent attention) for gfx1250 / MI455X — wave32 + WMMA.
//   B=4, S=2048, D=2048, H=16, HD=128, RD=64, QL=KL=512.
// All GEMMs + attention on v_wmma_f32_16x16x32_f16 with 32-byte vector operand
// loads (weights and kv pre-swizzled into B-tile order). Flash-style attention
// over the 512-wide latent. Q tiles are staged into LDS via the CDNA5 Tensor
// Data Mover (tensor_load_to_lds + s_wait_tensorcnt) when available.
// ---------------------------------------------------------------------------

typedef __attribute__((ext_vector_type(16))) _Float16 v16h;
typedef __attribute__((ext_vector_type(8)))  _Float16 v8h;
typedef __attribute__((ext_vector_type(8)))  float    v8f;
typedef __attribute__((ext_vector_type(4)))  unsigned int u32x4;
typedef __attribute__((ext_vector_type(8)))  int      i32x8;
typedef __attribute__((ext_vector_type(4)))  int      i32x4;

#if defined(__has_builtin)
#  if __has_builtin(__builtin_amdgcn_tensor_load_to_lds) && \
      __has_builtin(__builtin_amdgcn_s_wait_tensorcnt)
#    define HAVE_TDM 1
#  endif
#endif
#ifndef HAVE_TDM
#  define HAVE_TDM 0
#endif

namespace {
constexpr int kB  = 4;
constexpr int kS  = 2048;
constexpr int kD  = 2048;
constexpr int kH  = 16;
constexpr int kHD = 128;
constexpr int kRD = 64;
constexpr int kQL = 512;
constexpr int kKL = 512;
constexpr long kBS = (long)kB * kS;                 // 8192 token rows
constexpr float kScale = 0.07216878364870322f;      // 1/sqrt(HD+RD)
constexpr float kLog10000 = 9.210340371976184f;
}

// ---------------- WMMA operand loaders (CDNA5 wave32 layouts) ---------------
static __device__ inline v16h load_a16(const _Float16* p, int ldm, int lane) {
  const int m  = lane & 15;
  const int kb = (lane >> 4) * 16;
  const _Float16* row = p + (long)m * ldm + kb;
  v8h lo = *(const v8h*)row;
  v8h hi = *(const v8h*)(row + 8);
  return __builtin_shufflevector(lo, hi, 0, 1, 2, 3, 4, 5, 6, 7,
                                 8, 9, 10, 11, 12, 13, 14, 15);
}

static __device__ inline v16h load_b16_rows(const _Float16* p, long sn, int lane) {
  return *(const v16h*)(p + (long)(lane & 15) * sn + (lane >> 4) * 16);
}

static __device__ inline v16h load_b16_sw(const _Float16* tiles, long tile, int lane) {
  return *(const v16h*)(tiles + (tile * 32 + lane) * 16);
}

// RA = reuse-A hint: set when the previous WMMA is identical with the same A.
template <bool RA>
static __device__ inline v8f wmma_f16(v16h a, v16h b, v8f c) {
  return __builtin_amdgcn_wmma_f32_16x16x32_f16(false, a, false, b, (short)0, c,
                                                RA, false);
}

// --------------------- Tensor Data Mover 2D tile load -----------------------
#if HAVE_TDM
// D# per CDNA5 ISA. 2D tile: groups 2/3 zero.
// Dims/strides in elements (data_size = 2 bytes).
static __device__ inline void tdm_load_2d(unsigned lds_addr, const void* gaddr,
                                          unsigned tensor_d0, unsigned tensor_d1,
                                          unsigned stride_d0, unsigned tile_d0,
                                          unsigned tile_d1) {
  const unsigned long long ga = (unsigned long long)(size_t)gaddr;
  u32x4 g0;
  g0[0] = 1u;                                            // count=1 (user D#)
  g0[1] = lds_addr;                                      // LDS byte address
  g0[2] = (unsigned)(ga & 0xffffffffu);                  // global addr lo
  g0[3] = (unsigned)((ga >> 32) & 0x01ffffffu) | (2u << 30);  // hi | type=2
  i32x8 g1;
  g1[0] = (int)(1u << 16);                               // data_size = 2 bytes
  g1[1] = (int)((tensor_d0 & 0xffffu) << 16);            // tensor_dim0 lo16
  g1[2] = (int)((tensor_d0 >> 16) | ((tensor_d1 & 0xffffu) << 16));
  g1[3] = (int)((tensor_d1 >> 16) | (tile_d0 << 16));    // | tile_dim0
  g1[4] = (int)(tile_d1 & 0xffffu);                      // tile_dim1 (dim2=0)
  g1[5] = (int)stride_d0;                                // tensor_dim0_stride
  g1[6] = 0;
  g1[7] = 0;
  i32x4 z4 = {0, 0, 0, 0};
#if defined(__clang_major__) && __clang_major__ >= 23
  i32x8 z8 = {0, 0, 0, 0, 0, 0, 0, 0};
  __builtin_amdgcn_tensor_load_to_lds(g0, g1, z4, z4, z8, 0);
#else
  __builtin_amdgcn_tensor_load_to_lds(g0, g1, z4, z4, 0);
#endif
}

// Flat address of a __shared__ object truncates to the LDS byte address.
static __device__ inline unsigned lds_addr_of(const void* p) {
  return (unsigned)(unsigned long long)(size_t)p;
}
#endif

// ------------------------------ fp32 -> fp16 --------------------------------
__global__ __launch_bounds__(256) void cvt_f32_f16(const float* __restrict__ in,
                                                   _Float16* __restrict__ out,
                                                   long n) {
  long i = (long)blockIdx.x * blockDim.x + threadIdx.x;
  const long stride = (long)gridDim.x * blockDim.x;
  for (; i < n; i += stride) out[i] = (_Float16)in[i];
}

// --------------------- B-operand swizzler (+ fp16 convert) ------------------
template <typename T>
__global__ __launch_bounds__(256) void swizzle_b(
    const T* __restrict__ src, long sk, long sn, long head_src_stride,
    _Float16* __restrict__ dst, long head_dst_stride, int K, int N) {
  const int h = blockIdx.y;
  const long ktc = K / 32;
  const long total = (long)(N / 16) * ktc * 32;
  const long idx = (long)blockIdx.x * 256 + threadIdx.x;
  if (idx >= total) return;
  const int  lane = (int)(idx & 31);
  const long tile = idx >> 5;
  const long kt = tile % ktc, nt = tile / ktc;
  const long n  = nt * 16 + (lane & 15);
  const long k0 = kt * 32 + (lane >> 4) * 16;
  const T* s = src + (long)h * head_src_stride;
  v16h o;
#pragma unroll
  for (int e = 0; e < 16; ++e) o[e] = (_Float16)s[(k0 + e) * sk + n * sn];
  *(v16h*)(dst + (long)h * head_dst_stride + idx * 16) = o;
}

// ------------------------------- RMSNorm ------------------------------------
__global__ __launch_bounds__(128) void rmsnorm_rows(const float* __restrict__ in,
                                                    const float* __restrict__ gamma,
                                                    _Float16* __restrict__ out,
                                                    int L) {
  const long row = blockIdx.x;
  const float* r = in + row * (long)L;
  __shared__ float red[128];
  float s = 0.f;
  for (int i = threadIdx.x; i < L; i += 128) { float v = r[i]; s += v * v; }
  red[threadIdx.x] = s;
  __syncthreads();
#pragma unroll
  for (int w = 64; w > 0; w >>= 1) {
    if (threadIdx.x < (unsigned)w) red[threadIdx.x] += red[threadIdx.x + w];
    __syncthreads();
  }
  const float rs = rsqrtf(red[0] / (float)L + 1e-6f);
  for (int i = threadIdx.x; i < L; i += 128)
    out[row * (long)L + i] = (_Float16)(r[i] * rs * gamma[i]);
}

// --------------------------------- RoPE -------------------------------------
__global__ __launch_bounds__(256) void rope_rot(const float* __restrict__ in,
                                                _Float16* __restrict__ out,
                                                const int* __restrict__ start_pos,
                                                int nH) {
  const long idx = (long)blockIdx.x * blockDim.x + threadIdx.x;
  const long total = kBS * (long)nH * 32;
  if (idx >= total) return;
  const int  j   = (int)(idx & 31);
  const int  h   = (int)((idx >> 5) % nH);
  const long row = idx / ((long)nH * 32);
  const int  s   = (int)(row % kS);
  const float pos = (float)(start_pos[0] + s);
  const float inv = __expf(-(float)j * (kLog10000 / 32.f));
  const float ang = pos * inv;
  const float c = __cosf(ang), sn = __sinf(ang);
  const long base = (row * nH + h) * (long)kRD;
  const float x1 = in[base + j];
  const float x2 = in[base + j + 32];
  out[base + j]      = (_Float16)(x1 * c - x2 * sn);
  out[base + j + 32] = (_Float16)(x1 * sn + x2 * c);
}

// ------------------------- WMMA GEMM (swizzled B) ---------------------------
template <int NT>
__global__ __launch_bounds__(128) void gemm_wmma_sw(
    const _Float16* __restrict__ A, long lda,
    const _Float16* __restrict__ Bsw,
    const float* __restrict__ bias,
    float* __restrict__ Cf, _Float16* __restrict__ Ch, long ldc, int K) {
  const int lane = threadIdx.x & 31;
  const int wave = threadIdx.x >> 5;
  const long m0 = (long)blockIdx.y * 64 + (long)wave * 16;
  const long n0 = (long)blockIdx.x * (NT * 16);
  const long ktc = K >> 5;
  const _Float16* Abase = A + m0 * lda;
  const _Float16* Btile = Bsw + ((n0 / 16) * ktc * 32 + lane) * 16;

  v8f acc[NT];
#pragma unroll
  for (int t = 0; t < NT; ++t) { v8f z = {}; acc[t] = z; }

  for (int k = 0; k < K; k += 32) {
    if (k + 128 < K)  // prefetch hint -> global_prefetch_b8
      __builtin_prefetch((const void*)(Abase + k + 128), 0, 1);
    v16h a = load_a16(Abase + k, (int)lda, lane);
    {
      v16h b = *(const v16h*)(Btile + (long)(k >> 5) * (32 * 16));
      acc[0] = wmma_f16<false>(a, b, acc[0]);
    }
#pragma unroll
    for (int t = 1; t < NT; ++t) {
      v16h b = *(const v16h*)(Btile + ((long)t * ktc + (k >> 5)) * (32 * 16));
      acc[t] = wmma_f16<true>(a, b, acc[t]);  // same A as previous WMMA
    }
  }
  const int n  = lane & 15;
  const int mh = (lane >> 4) * 8;
#pragma unroll
  for (int t = 0; t < NT; ++t) {
    const long nc = n0 + t * 16 + n;
    const float bv = bias ? bias[nc] : 0.f;
#pragma unroll
    for (int r = 0; r < 8; ++r) {
      const float val = acc[t][r] + bv;
      const long idx = (m0 + r + mh) * ldc + nc;
      if (Cf) Cf[idx] = val;
      if (Ch) Ch[idx] = (_Float16)val;
    }
  }
}

// ----------------------- Flash MLA attention kernel -------------------------
__global__ __launch_bounds__(128) void mla_flash(
    const _Float16* __restrict__ qabs,   // [BS, H, KL]
    const _Float16* __restrict__ qrope,  // [BS, H, RD]
    const _Float16* __restrict__ kv,     // [BS, KL] row-major
    const _Float16* __restrict__ kvsw,   // [BS/32 bands][32 tiles][32][16]
    const _Float16* __restrict__ krope,  // [BS, RD]
    const _Float16* __restrict__ wvusw,  // swizzled K=512,N=2048
    const float* __restrict__ gates,     // [H]
    _Float16* __restrict__ ctx)          // [BS, H*HD]
{
  const int tid  = threadIdx.x;
  const int lane = tid & 31;
  const int wave = tid >> 5;
  const int q0   = blockIdx.x * 16;
  const int h    = blockIdx.y;
  const long rowbase = (long)blockIdx.z * kS;

  __shared__ __align__(32) _Float16 sQ [16 * kKL];   // reused as acc stage
  __shared__ __align__(32) _Float16 sQr[16 * kRD];
  __shared__ __align__(32) _Float16 sP [16 * 32];
  __shared__ float sS[16 * 32];
  __shared__ float sM[16], sL[16], sFac[16];

#if HAVE_TDM
  // Stage q_abs / q_rope tiles with the Tensor Data Mover (async DMA).
  if (wave == 0) {
    tdm_load_2d(lds_addr_of(sQ),
                qabs + ((rowbase + q0) * kH + h) * (long)kKL,
                kKL, 16, (unsigned)(kH * kKL), kKL, 16);
  } else if (wave == 1) {
    tdm_load_2d(lds_addr_of(sQr),
                qrope + ((rowbase + q0) * kH + h) * (long)kRD,
                kRD, 16, (unsigned)(kH * kRD), kRD, 16);
  }
  __builtin_amdgcn_s_wait_tensorcnt((short)0);  // no-op for waves 2..3
#else
  for (int i = tid; i < 16 * kKL / 8; i += 128) {
    const int r = i / (kKL / 8), c = i % (kKL / 8);
    ((uint4*)sQ)[i] =
        ((const uint4*)(qabs + ((rowbase + q0 + r) * kH + h) * (long)kKL))[c];
  }
  for (int i = tid; i < 16 * kRD / 8; i += 128) {
    const int r = i / (kRD / 8), c = i % (kRD / 8);
    ((uint4*)sQr)[i] =
        ((const uint4*)(qrope + ((rowbase + q0 + r) * kH + h) * (long)kRD))[c];
  }
#endif
  if (tid < 16) { sM[tid] = -3.0e38f; sL[tid] = 0.f; }

  v8f acc[8];
#pragma unroll
  for (int i = 0; i < 8; ++i) { v8f z = {}; acc[i] = z; }
  const int n_base = wave * 128;  // this wave's 128 latent columns
  __syncthreads();

  for (int t0 = 0; t0 < q0 + 16; t0 += 32) {
#pragma unroll
    for (int i = 0; i < 4; ++i) sS[tid + i * 128] = 0.f;
    __syncthreads();

    // ---- partial scores (B direct from global: contiguous kv^T slices) ----
    const _Float16* kvb = kv + (rowbase + t0) * (long)kKL;
    v8f s0 = {}, s1 = {};
#pragma unroll
    for (int kk = 0; kk < 128; kk += 32) {
      const int k = n_base + kk;
      v16h a  = load_a16(sQ + k, kKL, lane);
      v16h b0 = load_b16_rows(kvb + k, kKL, lane);              // t 0..15
      v16h b1 = load_b16_rows(kvb + 16 * kKL + k, kKL, lane);   // t 16..31
      s0 = wmma_f16<false>(a, b0, s0);
      s1 = wmma_f16<true>(a, b1, s1);   // same A as previous WMMA
    }
    if (wave < 2) {  // rope contribution: wave0 -> k 0..31, wave1 -> k 32..63
      const int kk = wave * 32;
      const _Float16* krb = krope + (rowbase + t0) * (long)kRD + kk;
      v16h a  = load_a16(sQr + kk, kRD, lane);
      v16h b0 = load_b16_rows(krb, kRD, lane);
      v16h b1 = load_b16_rows(krb + 16 * kRD, kRD, lane);
      s0 = wmma_f16<false>(a, b0, s0);
      s1 = wmma_f16<true>(a, b1, s1);
    }
    {
      const int n = lane & 15, mh = (lane >> 4) * 8;
#pragma unroll
      for (int r = 0; r < 8; ++r) {
        atomicAdd(&sS[(r + mh) * 32 + n],      s0[r]);
        atomicAdd(&sS[(r + mh) * 32 + 16 + n], s1[r]);
      }
    }
    __syncthreads();

    // ---- online softmax (one thread per q row) ----
    if (tid < 16) {
      const int qrow = q0 + tid;
      const float m_old = sM[tid];
      float mx = m_old;
      float sc[32];
#pragma unroll
      for (int j = 0; j < 32; ++j) {
        const float v = (t0 + j <= qrow) ? sS[tid * 32 + j] * kScale : -3.0e38f;
        sc[j] = v;
        mx = fmaxf(mx, v);
      }
      const float fac = __expf(m_old - mx);
      float l = sL[tid] * fac;
#pragma unroll
      for (int j = 0; j < 32; ++j) {
        const float p = (t0 + j <= qrow) ? __expf(sc[j] - mx) : 0.f;
        l += p;
        sP[tid * 32 + j] = (_Float16)p;
      }
      sM[tid] = mx; sL[tid] = l; sFac[tid] = fac;
    }
    __syncthreads();

    // ---- rescale acc, then acc += p @ kv_chunk (pre-swizzled B tiles) ----
    {
      const int mh = (lane >> 4) * 8;
      float fr[8];
#pragma unroll
      for (int r = 0; r < 8; ++r) fr[r] = sFac[r + mh];
#pragma unroll
      for (int t = 0; t < 8; ++t)
#pragma unroll
        for (int r = 0; r < 8; ++r) acc[t][r] *= fr[r];
      v16h a = load_a16(sP, 32, lane);  // p[16,32], K = 32 t-rows
      const _Float16* band = kvsw + (rowbase + t0) * (long)kKL;  // 32 tiles/band
      acc[0] = wmma_f16<false>(a, load_b16_sw(band, wave * 8, lane), acc[0]);
#pragma unroll
      for (int t = 1; t < 8; ++t) {
        v16h b = load_b16_sw(band, wave * 8 + t, lane);
        acc[t] = wmma_f16<true>(a, b, acc[t]);  // same A as previous WMMA
      }
    }
    __syncthreads();
  }

  // ---- epilogue: normalize, restage through LDS (reuse sQ) ----
  {
    const int n = lane & 15, mh = (lane >> 4) * 8;
    float ln[8];
#pragma unroll
    for (int r = 0; r < 8; ++r) ln[r] = 1.f / sL[r + mh];
#pragma unroll
    for (int t = 0; t < 8; ++t)
#pragma unroll
      for (int r = 0; r < 8; ++r)
        sQ[(r + mh) * kKL + n_base + t * 16 + n] = (_Float16)(acc[t][r] * ln[r]);
  }
  __syncthreads();

  // ---- out = attn_latent @ w_uv[h]^T (swizzled tiles; 32 d-cols per wave) --
  v8f o0 = {}, o1 = {};
  const long ktc = kKL / 32;                       // 16
  const long nt0 = (long)h * 8 + wave * 2;         // global n-tile index
#pragma unroll
  for (int kk = 0; kk < kKL; kk += 32) {
    v16h a  = load_a16(sQ + kk, kKL, lane);
    v16h b0 = load_b16_sw(wvusw, nt0 * ktc + (kk >> 5), lane);
    v16h b1 = load_b16_sw(wvusw, (nt0 + 1) * ktc + (kk >> 5), lane);
    o0 = wmma_f16<false>(a, b0, o0);
    o1 = wmma_f16<true>(a, b1, o1);
  }
  const float gate = 1.f / (1.f + __expf(-gates[h]));
  {
    const int n = lane & 15, mh = (lane >> 4) * 8;
#pragma unroll
    for (int r = 0; r < 8; ++r) {
      const long row = rowbase + q0 + r + mh;
      ctx[row * (kH * kHD) + h * kHD + wave * 32 + n]      = (_Float16)(o0[r] * gate);
      ctx[row * (kH * kHD) + h * kHD + wave * 32 + 16 + n] = (_Float16)(o1[r] * gate);
    }
  }
}

// ------------------------------- launcher -----------------------------------
extern "C" void kernel_launch(void* const* d_in, const int* in_sizes, int n_in,
                              void* d_out, int out_size, void* d_ws, size_t ws_size,
                              hipStream_t stream) {
  (void)in_sizes; (void)n_in; (void)out_size; (void)ws_size;
  const float* x     = (const float*)d_in[0];
  const int*   stp   = (const int*)  d_in[2];
  const float* w_qd  = (const float*)d_in[3];
  const float* b_qd  = (const float*)d_in[4];
  const float* g_q   = (const float*)d_in[5];
  const float* w_qn  = (const float*)d_in[6];
  const float* b_qn  = (const float*)d_in[7];
  const float* w_qr  = (const float*)d_in[8];
  const float* b_qr  = (const float*)d_in[9];
  const float* w_kvd = (const float*)d_in[10];
  const float* b_kvd = (const float*)d_in[11];
  const float* g_kv  = (const float*)d_in[12];
  const float* w_ku  = (const float*)d_in[13];
  const float* w_vu  = (const float*)d_in[15];
  const float* w_kr  = (const float*)d_in[17];
  const float* b_kr  = (const float*)d_in[18];
  const float* gates = (const float*)d_in[19];
  const float* w_o   = (const float*)d_in[20];
  const float* b_o   = (const float*)d_in[21];
  float* outp = (float*)d_out;

  size_t off = 0;
  auto alloc = [&](size_t bytes) {
    void* p = (char*)d_ws + off;
    off += (bytes + 255) & ~(size_t)255;
    return p;
  };
  _Float16* x_h     = (_Float16*)alloc(kBS * kD * 2);
  _Float16* qc_h    = (_Float16*)alloc(kBS * kQL * 2);
  _Float16* kv_h    = (_Float16*)alloc(kBS * kKL * 2);
  _Float16* qn_h    = (_Float16*)alloc(kBS * kD * 2);
  _Float16* qr_h    = (_Float16*)alloc(kBS * (size_t)kH * kRD * 2);
  _Float16* kr_h    = (_Float16*)alloc(kBS * kRD * 2);
  _Float16* qabs_h  = (_Float16*)alloc(kBS * (size_t)kH * kKL * 2);
  _Float16* ctx_h   = (_Float16*)alloc(kBS * kD * 2);
  _Float16* kv_sw   = (_Float16*)alloc(kBS * kKL * 2);
  _Float16* wqd_sw  = (_Float16*)alloc((size_t)kQL * kD * 2);
  _Float16* wqn_sw  = (_Float16*)alloc((size_t)kD * kQL * 2);
  _Float16* wqr_sw  = (_Float16*)alloc((size_t)kH * kRD * kQL * 2);
  _Float16* wkvd_sw = (_Float16*)alloc((size_t)kKL * kD * 2);
  _Float16* wku_sw  = (_Float16*)alloc((size_t)kD * kKL * 2);
  _Float16* wvu_sw  = (_Float16*)alloc((size_t)kD * kKL * 2);
  _Float16* wkr_sw  = (_Float16*)alloc((size_t)kRD * kD * 2);
  _Float16* wo_sw   = (_Float16*)alloc((size_t)kD * kD * 2);
  float*    tmpf    = (float*)   alloc(kBS * 1024 * 4);

  cvt_f32_f16<<<4096, 256, 0, stream>>>(x, x_h, kBS * kD);

  auto swz32 = [&](const float* src, long sk, long sn, long hs,
                   _Float16* dst, long hd, int K, int N, int heads) {
    const long total = (long)(N / 16) * (K / 32) * 32;
    dim3 grid((unsigned)((total + 255) / 256), (unsigned)heads);
    swizzle_b<float><<<grid, 256, 0, stream>>>(src, sk, sn, hs, dst, hd, K, N);
  };
  swz32(w_qd,  1, kD,  0, wqd_sw,  0, kD,  kQL, 1);
  swz32(w_qn,  1, kQL, 0, wqn_sw,  0, kQL, kD,  1);
  swz32(w_qr,  1, kQL, 0, wqr_sw,  0, kQL, kH * kRD, 1);
  swz32(w_kvd, 1, kD,  0, wkvd_sw, 0, kD,  kKL, 1);
  swz32(w_ku,  kKL, 1, (long)kHD * kKL, wku_sw,
        (long)(kKL / 16) * (kHD / 32) * 512, kHD, kKL, kH);
  swz32(w_vu,  1, kKL, 0, wvu_sw,  0, kKL, kD, 1);
  swz32(w_kr,  1, kD,  0, wkr_sw,  0, kD,  kRD, 1);
  swz32(w_o,   1, kD,  0, wo_sw,   0, kD,  kD,  1);

  auto gemm8 = [&](const _Float16* A, long lda, const _Float16* Bsw,
                   const float* bias, float* Cf, _Float16* Ch, long ldc,
                   int N, int K) {
    dim3 grid((unsigned)(N / 128), (unsigned)(kBS / 64));
    gemm_wmma_sw<8><<<grid, 128, 0, stream>>>(A, lda, Bsw, bias, Cf, Ch, ldc, K);
  };

  // q path: down -> rmsnorm -> nope_up / rope_up(+rope)
  gemm8(x_h, kD, wqd_sw, b_qd, tmpf, nullptr, kQL, kQL, kD);
  rmsnorm_rows<<<(unsigned)kBS, 128, 0, stream>>>(tmpf, g_q, qc_h, kQL);
  gemm8(qc_h, kQL, wqn_sw, b_qn, nullptr, qn_h, kD, kD, kQL);
  gemm8(qc_h, kQL, wqr_sw, b_qr, tmpf, nullptr, (long)kH * kRD, kH * kRD, kQL);
  rope_rot<<<(unsigned)(kBS * kH * 32 / 256), 256, 0, stream>>>(tmpf, qr_h, stp, kH);

  // kv path: down -> rmsnorm -> band-swizzle ; k_rope: x @ w_kr^T -> rope
  gemm8(x_h, kD, wkvd_sw, b_kvd, tmpf, nullptr, kKL, kKL, kD);
  rmsnorm_rows<<<(unsigned)kBS, 128, 0, stream>>>(tmpf, g_kv, kv_h, kKL);
  {  // kv_sw: per 32-row band, K=32, N=512, element (k,n)=kv[band*32+k][n]
    const long total = (long)(kKL / 16) * 1 * 32;
    dim3 grid((unsigned)((total + 255) / 256), (unsigned)(kBS / 32));
    swizzle_b<_Float16><<<grid, 256, 0, stream>>>(
        kv_h, kKL, 1, 32L * kKL, kv_sw, 32L * kKL, 32, kKL);
  }
  {
    dim3 grid(1, (unsigned)(kBS / 64));
    gemm_wmma_sw<4><<<grid, 128, 0, stream>>>(x_h, kD, wkr_sw, b_kr, tmpf,
                                              nullptr, kRD, kD);
  }
  rope_rot<<<(unsigned)(kBS * 32 / 256), 256, 0, stream>>>(tmpf, kr_h, stp, 1);

  // q_abs[b,s,h,:] = q_nope[b,s,h,:] @ w_uk[h]  (per-head GEMM, K=HD=128)
  const long wku_head = (long)(kKL / 16) * (kHD / 32) * 512;
  for (int h = 0; h < kH; ++h) {
    dim3 grid((unsigned)(kKL / 128), (unsigned)(kBS / 64));
    gemm_wmma_sw<8><<<grid, 128, 0, stream>>>(
        qn_h + (long)h * kHD, kD, wku_sw + h * wku_head, nullptr,
        nullptr, qabs_h + (long)h * kKL, (long)kH * kKL, kHD);
  }

  // flash attention + per-head out projection + gate -> ctx fp16
  {
    dim3 grid(kS / 16, kH, kB);
    mla_flash<<<grid, 128, 0, stream>>>(qabs_h, qr_h, kv_h, kv_sw, kr_h,
                                        wvu_sw, gates, ctx_h);
  }

  // final projection: out = ctx @ w_out^T + b_out (fp32 result)
  gemm8(ctx_h, kD, wo_sw, b_o, outp, nullptr, kD, kD, kD);
}
